// RWKV_Tmix_Dual_66915590472469
// MI455X (gfx1250) — matrix operand order repaired and checked by reading, not compiled, not run on hardware
//
#include <hip/hip_runtime.h>
#include <hip/hip_bf16.h>

// ---------------------------------------------------------------------------
// RWKV-7 Tmix forward for MI455X (gfx1250), wave32 + WMMA f16->f32.
// B=4 T=1024 C=1024 H=16 N=64 ; M = B*T = 4096 rows.
// ---------------------------------------------------------------------------

#define Mrows 4096
#define Cch   1024
#define Hh    16
#define Nn    64
#define Tt    1024
#define Bb    4

typedef __attribute__((ext_vector_type(16))) _Float16 v16h;
typedef __attribute__((ext_vector_type(8)))  _Float16 v8h;
typedef __attribute__((ext_vector_type(8)))  float    v8f;

// ---------------------------------------------------------------------------
// Token-shift mix: xx = shift(x) - x ; six mixed tensors -> f16
// ---------------------------------------------------------------------------
__global__ __launch_bounds__(256)
void mix_kernel(const float* __restrict__ x,
                const float* __restrict__ mr, const float* __restrict__ mw,
                const float* __restrict__ mk, const float* __restrict__ mv,
                const float* __restrict__ ma, const float* __restrict__ mg,
                _Float16* __restrict__ oxr, _Float16* __restrict__ oxw,
                _Float16* __restrict__ oxk, _Float16* __restrict__ oxv,
                _Float16* __restrict__ oxa, _Float16* __restrict__ oxg)
{
    size_t idx = (size_t)blockIdx.x * blockDim.x + threadIdx.x;
    int c = (int)(idx & (Cch - 1));
    int t = (int)((idx >> 10) & (Tt - 1));
    float xc = x[idx];
    float xp = (t == 0) ? 0.0f : x[idx - Cch];
    float xx = xp - xc;
    oxr[idx] = (_Float16)(xc + xx * mr[c]);
    oxw[idx] = (_Float16)(xc + xx * mw[c]);
    oxk[idx] = (_Float16)(xc + xx * mk[c]);
    oxv[idx] = (_Float16)(xc + xx * mv[c]);
    oxa[idx] = (_Float16)(xc + xx * ma[c]);
    oxg[idx] = (_Float16)(xc + xx * mg[c]);
}

// ---------------------------------------------------------------------------
// Weight transpose + convert: src (K x N) f32 row-major -> dst (Npad x Kpad)
// f16 row-major ("Bt[n][k]"), zero padding keeps padded GEMMs exact.
// ---------------------------------------------------------------------------
__global__ __launch_bounds__(256)
void wtrans_kernel(const float* __restrict__ src, _Float16* __restrict__ dst,
                   int K, int Ncols, int Npad, int Kpad)
{
    size_t idx = (size_t)blockIdx.x * blockDim.x + threadIdx.x;
    if (idx >= (size_t)Npad * Kpad) return;
    int n = (int)(idx / Kpad);
    int k = (int)(idx % Kpad);
    float v = (n < Ncols && k < K) ? src[(size_t)k * Ncols + n] : 0.0f;
    dst[idx] = (_Float16)v;
}

// ---------------------------------------------------------------------------
// Activation + f32 -> f16 convert (mode 0: copy, 1: tanh, 2: sigmoid)
// ---------------------------------------------------------------------------
__global__ __launch_bounds__(256)
void act_kernel(const float* __restrict__ src, _Float16* __restrict__ dst,
                int n, int mode)
{
    int idx = blockIdx.x * blockDim.x + threadIdx.x;
    if (idx >= n) return;
    float v = src[idx];
    if (mode == 1)      v = tanhf(v);
    else if (mode == 2) v = 1.0f / (1.0f + expf(-v));
    dst[idx] = (_Float16)v;
}

// ---------------------------------------------------------------------------
// WMMA GEMM: C(MxN,f32) = A(MxK,f16,lda) * Bt(NxK,f16,ldb)^T
// Block = 128 threads = 4 waves; block tile 128(M) x 64(N); each wave owns a
// 32x64 strip: 2 A-row strips x 4 col tiles = 8 accumulators, 8 WMMAs per
// k-step reusing each B fragment twice (~409 FLOP/B from cache).
//
// Software pipeline: fragments for block k+32 are fetched before the 8 WMMAs
// on block k. The final k-block is peeled into an epilogue so the in-loop
// fetch index stays affine (immediate-offset addressing, no cselect chains).
//
// Fragment layouts used here (contiguous K, not the ISA table order; A and B match), g = lane/16:
//  A 16x32: lane row m=lane%16; two 16B loads covering contiguous K 16g..16g+15.
//  B 32x16: lane col n=lane%16; one contiguous 32B load from Bt[n][k+16g].
// ---------------------------------------------------------------------------
__device__ __forceinline__ v16h load_afrag(const _Float16* p)
{
    v8h lo = *(const v8h*)(p);
    v8h hi = *(const v8h*)(p + 8);
    v16h f;
#pragma unroll
    for (int j = 0; j < 8; ++j) { f[j] = lo[j]; f[8 + j] = hi[j]; }
    return f;
}

#define WMMA_F16(Af, Bf, Acc) \
    __builtin_amdgcn_wmma_f32_16x16x32_f16(false, (Af), false, (Bf), (short)0, \
                                           (Acc), false, false)

__global__ __launch_bounds__(128)
void gemm_wmma_f16(const _Float16* __restrict__ A, const _Float16* __restrict__ Bt,
                   float* __restrict__ C, int K, int lda, int ldb, int ldc)
{
    const int lane = threadIdx.x & 31;
    const int wave = threadIdx.x >> 5;
    const int g    = lane >> 4;
    const int l16  = lane & 15;
    const int row0 = blockIdx.y * 128 + wave * 32;   // two 16-row strips
    const int col0 = blockIdx.x * 64;

    const _Float16* ar0 = A + (size_t)(row0 +      l16) * lda + 16 * g;
    const _Float16* ar1 = A + (size_t)(row0 + 16 + l16) * lda + 16 * g;
    const _Float16* b0 = Bt + (size_t)(col0 +  0 + l16) * ldb + 16 * g;
    const _Float16* b1 = Bt + (size_t)(col0 + 16 + l16) * ldb + 16 * g;
    const _Float16* b2 = Bt + (size_t)(col0 + 32 + l16) * ldb + 16 * g;
    const _Float16* b3 = Bt + (size_t)(col0 + 48 + l16) * ldb + 16 * g;

    v8f acc00 = {}, acc01 = {}, acc02 = {}, acc03 = {};
    v8f acc10 = {}, acc11 = {}, acc12 = {}, acc13 = {};

    // prologue: fragments for k-block 0
    v16h a0c = load_afrag(ar0);
    v16h a1c = load_afrag(ar1);
    v16h bc0 = *(const v16h*)(b0);
    v16h bc1 = *(const v16h*)(b1);
    v16h bc2 = *(const v16h*)(b2);
    v16h bc3 = *(const v16h*)(b3);

    int kb = 0;
    for (; kb + 32 < K; kb += 32) {
        const int kn = kb + 32;                       // affine next index
        __builtin_prefetch(ar0 + kn + 256, 0, 1);     // global_prefetch_b8
        v16h a0n = load_afrag(ar0 + kn);
        v16h a1n = load_afrag(ar1 + kn);
        v16h nb0 = *(const v16h*)(b0 + kn);
        v16h nb1 = *(const v16h*)(b1 + kn);
        v16h nb2 = *(const v16h*)(b2 + kn);
        v16h nb3 = *(const v16h*)(b3 + kn);

        acc00 = WMMA_F16(a0c, bc0, acc00);
        acc01 = WMMA_F16(a0c, bc1, acc01);
        acc02 = WMMA_F16(a0c, bc2, acc02);
        acc03 = WMMA_F16(a0c, bc3, acc03);
        acc10 = WMMA_F16(a1c, bc0, acc10);
        acc11 = WMMA_F16(a1c, bc1, acc11);
        acc12 = WMMA_F16(a1c, bc2, acc12);
        acc13 = WMMA_F16(a1c, bc3, acc13);

        a0c = a0n; a1c = a1n;
        bc0 = nb0; bc1 = nb1; bc2 = nb2; bc3 = nb3;
    }

    // epilogue: last k-block (fragments already resident)
    acc00 = WMMA_F16(a0c, bc0, acc00);
    acc01 = WMMA_F16(a0c, bc1, acc01);
    acc02 = WMMA_F16(a0c, bc2, acc02);
    acc03 = WMMA_F16(a0c, bc3, acc03);
    acc10 = WMMA_F16(a1c, bc0, acc10);
    acc11 = WMMA_F16(a1c, bc1, acc11);
    acc12 = WMMA_F16(a1c, bc2, acc12);
    acc13 = WMMA_F16(a1c, bc3, acc13);

    // C/D layout: VGPR i of lane L -> row = base + i + 8*(L/16), col = col0+L%16
#pragma unroll
    for (int i = 0; i < 8; ++i) {
        float* cp0 = C + (size_t)(row0 + 8 * g + i) * ldc + col0 + l16;
        cp0[ 0] = acc00[i];
        cp0[16] = acc01[i];
        cp0[32] = acc02[i];
        cp0[48] = acc03[i];
        float* cp1 = C + (size_t)(row0 + 16 + 8 * g + i) * ldc + col0 + l16;
        cp1[ 0] = acc10[i];
        cp1[16] = acc11[i];
        cp1[32] = acc12[i];
        cp1[48] = acc13[i];
    }
}

// ---------------------------------------------------------------------------
// Fused pre-scan elementwise. One block per token (256 thr x 4 ch).
// ---------------------------------------------------------------------------
__global__ __launch_bounds__(256)
void prescan_kernel(const float* __restrict__ wpre, const float* __restrict__ apre,
                    const float* __restrict__ vpre,
                    const float* __restrict__ w0, const float* __restrict__ a0,
                    const float* __restrict__ v0,
                    const float* __restrict__ k_k, const float* __restrict__ k_a,
                    const float* __restrict__ v_first,
                    float* __restrict__ kbuf, float* __restrict__ vbuf,
                    float* __restrict__ wdb, float* __restrict__ nkk,
                    float* __restrict__ bkk)
{
    size_t base = (size_t)blockIdx.x * Cch;
    int tid = threadIdx.x;
    int c0 = tid * 4;

    float kv[4], kkv[4];
    float ss = 0.0f;
#pragma unroll
    for (int u = 0; u < 4; ++u) {
        int c = c0 + u;
        float kc = kbuf[base + c];
        kv[u] = kc;
        float t = kc * k_k[c];
        kkv[u] = t;
        ss += t * t;
    }
    // head = tid/16 : sum-of-squares across each 16-lane segment (wave32)
    ss += __shfl_xor(ss, 1, 16);
    ss += __shfl_xor(ss, 2, 16);
    ss += __shfl_xor(ss, 4, 16);
    ss += __shfl_xor(ss, 8, 16);
    float inv = 1.0f / fmaxf(sqrtf(ss), 1e-12f);

#pragma unroll
    for (int u = 0; u < 4; ++u) {
        int c = c0 + u;
        float ac = 1.0f / (1.0f + expf(-(a0[c] + apre[base + c])));
        float z  = -(w0[c] + wpre[base + c]);
        float sp = (z > 20.0f) ? z : log1pf(expf(z));
        float w  = -sp - 0.5f;
        float wdv = expf(-expf(w));
        float sv = 1.0f / (1.0f + expf(-(v0[c] + vpre[base + c])));
        float vr = vbuf[base + c];
        float vf = vr + (v_first[base + c] - vr) * sv;
        float kkn = kkv[u] * inv;
        float kf  = kv[u] * (1.0f + (ac - 1.0f) * k_a[c]);
        wdb[base + c] = wdv;
        nkk[base + c] = -kkn;
        bkk[base + c] = kkn * ac;
        kbuf[base + c] = kf;
        vbuf[base + c] = vf;
    }
}

// ---------------------------------------------------------------------------
// Sequential delta-rule scan. One block per (b,h); 64 threads; lane i owns
// state row i (64 f32 in registers).
//
// Double-buffered LDS staging of (a,b,k,r) vectors for step t+1 via CDNA5
// async global->LDS copies (ASYNCcnt), hiding global latency behind the
// serial step-t math. LDS dest operand = low 32 bits of the generic shared
// address (ISA: LDS aperture addresses map to LDS by discarding upper bits).
// ---------------------------------------------------------------------------
__global__ __launch_bounds__(64)
void scan_kernel(const float* __restrict__ wd, const float* __restrict__ r,
                 const float* __restrict__ k, const float* __restrict__ v,
                 const float* __restrict__ na, const float* __restrict__ bb,
                 float* __restrict__ y)
{
    int b = blockIdx.x >> 4;
    int h = blockIdx.x & 15;
    int i = threadIdx.x;

    float S[Nn];
#pragma unroll
    for (int j = 0; j < Nn; ++j) S[j] = 0.0f;

    __shared__ float as_[2][Nn], bs_[2][Nn], ks_[2][Nn], rs_[2][Nn];

    const size_t base0 = (size_t)b * Tt * Cch + h * Nn + i;

    // prologue: step 0 via normal loads
    as_[0][i] = na[base0];
    bs_[0][i] = bb[base0];
    ks_[0][i] = k [base0];
    rs_[0][i] = r [base0];
    __syncthreads();

    int p = 0;
    for (int t = 0; t < Tt; ++t) {
        size_t base = base0 + (size_t)t * Cch;
        float wdi = wd[base];
        float vi  = v [base];

        if (t + 1 < Tt) {
            size_t nb = base + Cch;
            unsigned la = (unsigned)(size_t)&as_[p ^ 1][i];
            unsigned lb = (unsigned)(size_t)&bs_[p ^ 1][i];
            unsigned lk = (unsigned)(size_t)&ks_[p ^ 1][i];
            unsigned lr = (unsigned)(size_t)&rs_[p ^ 1][i];
            asm volatile("global_load_async_to_lds_b32 %0, %1, off"
                         :: "v"(la), "v"(na + nb) : "memory");
            asm volatile("global_load_async_to_lds_b32 %0, %1, off"
                         :: "v"(lb), "v"(bb + nb) : "memory");
            asm volatile("global_load_async_to_lds_b32 %0, %1, off"
                         :: "v"(lk), "v"(k + nb) : "memory");
            asm volatile("global_load_async_to_lds_b32 %0, %1, off"
                         :: "v"(lr), "v"(r + nb) : "memory");
        }

        float sa = 0.0f;
#pragma unroll
        for (int j = 0; j < Nn; ++j) sa += S[j] * as_[p][j];

        float yi = 0.0f;
#pragma unroll
        for (int j = 0; j < Nn; ++j) {
            float sj = S[j] * wdi + sa * bs_[p][j] + vi * ks_[p][j];
            S[j] = sj;
            yi += sj * rs_[p][j];
        }
        y[base] = yi;

        asm volatile("s_wait_asynccnt 0x0" ::: "memory");
        __syncthreads();
        p ^= 1;
    }
}

// ---------------------------------------------------------------------------
// GroupNorm (per head) + bonus term + gate -> (y*g) f16. EPS = 1e-5*64
// ---------------------------------------------------------------------------
__global__ __launch_bounds__(64)
void gn_post_kernel(const float* __restrict__ y, const float* __restrict__ r,
                    const float* __restrict__ k, const float* __restrict__ v,
                    const float* __restrict__ g, const float* __restrict__ rk,
                    const float* __restrict__ lnw, const float* __restrict__ lnb,
                    _Float16* __restrict__ yg)
{
    int tok = blockIdx.x >> 4;
    int h   = blockIdx.x & 15;
    int i   = threadIdx.x;
    size_t base = (size_t)tok * Cch + h * Nn;

    float yi = y[base + i];
    __shared__ float s1[Nn], s2[Nn];
    s1[i] = yi;
    s2[i] = yi * yi;
    __syncthreads();
    float sum = 0.0f, sq = 0.0f;
    for (int j = 0; j < Nn; ++j) { sum += s1[j]; sq += s2[j]; }
    float mean = sum * (1.0f / Nn);
    float var  = sq * (1.0f / Nn) - mean * mean;
    float yn = (yi - mean) * rsqrtf(var + 6.4e-4f);
    yn = yn * lnw[h * Nn + i] + lnb[h * Nn + i];

    float ri = r[base + i], ki = k[base + i], vi = v[base + i];
    __syncthreads();
    s1[i] = ri * ki * rk[h * Nn + i];
    __syncthreads();
    float dot = 0.0f;
    for (int j = 0; j < Nn; ++j) dot += s1[j];

    float yf = yn + dot * vi;
    yg[base + i] = (_Float16)(yf * g[base + i]);
}

// ---------------------------------------------------------------------------
// Host-side orchestration
// ---------------------------------------------------------------------------
extern "C" void kernel_launch(void* const* d_in, const int* in_sizes, int n_in,
                              void* d_out, int out_size, void* d_ws, size_t ws_size,
                              hipStream_t stream)
{
    (void)in_sizes; (void)n_in; (void)out_size; (void)ws_size;

    const float* x       = (const float*)d_in[0];
    const float* v_first = (const float*)d_in[1];
    const float* x_r = (const float*)d_in[2];
    const float* x_w = (const float*)d_in[3];
    const float* x_k = (const float*)d_in[4];
    const float* x_v = (const float*)d_in[5];
    const float* x_a = (const float*)d_in[6];
    const float* x_g = (const float*)d_in[7];
    const float* w0  = (const float*)d_in[8];
    const float* w1  = (const float*)d_in[9];
    const float* w2  = (const float*)d_in[10];
    const float* a0  = (const float*)d_in[11];
    const float* a1  = (const float*)d_in[12];
    const float* a2  = (const float*)d_in[13];
    const float* v0  = (const float*)d_in[14];
    const float* v1  = (const float*)d_in[15];
    const float* v2  = (const float*)d_in[16];
    const float* g1  = (const float*)d_in[17];
    const float* g2  = (const float*)d_in[18];
    const float* k_k = (const float*)d_in[19];
    const float* k_a = (const float*)d_in[20];
    const float* r_k = (const float*)d_in[21];
    const float* W_r = (const float*)d_in[22];
    const float* W_k = (const float*)d_in[23];
    const float* W_v = (const float*)d_in[24];
    const float* W_o = (const float*)d_in[25];
    const float* lnw = (const float*)d_in[26];
    const float* lnb = (const float*)d_in[27];

    const size_t MC = (size_t)Mrows * Cch;

    char* p = (char*)d_ws;
    auto carve = [&](size_t bytes) -> void* {
        void* q = (void*)p;
        p += (bytes + 255) & ~(size_t)255;
        return q;
    };

    _Float16* xr = (_Float16*)carve(MC * 2);
    _Float16* xw = (_Float16*)carve(MC * 2);
    _Float16* xk = (_Float16*)carve(MC * 2);
    _Float16* xv = (_Float16*)carve(MC * 2);
    _Float16* xa = (_Float16*)carve(MC * 2);
    _Float16* xg = (_Float16*)carve(MC * 2);

    _Float16* Wrt = (_Float16*)carve((size_t)Cch * Cch * 2);
    _Float16* Wkt = (_Float16*)carve((size_t)Cch * Cch * 2);
    _Float16* Wvt = (_Float16*)carve((size_t)Cch * Cch * 2);
    _Float16* Wot = (_Float16*)carve((size_t)Cch * Cch * 2);
    _Float16* w1t = (_Float16*)carve((size_t)64  * 1024 * 2);  // [64][1024]
    _Float16* w2t = (_Float16*)carve((size_t)1024 * 64  * 2);  // [1024][64]
    _Float16* a1t = (_Float16*)carve((size_t)64  * 1024 * 2);
    _Float16* a2t = (_Float16*)carve((size_t)1024 * 64  * 2);
    _Float16* v1t = (_Float16*)carve((size_t)64  * 1024 * 2);  // Npad 32->64
    _Float16* v2t = (_Float16*)carve((size_t)1024 * 64  * 2);  // Kpad 32->64
    _Float16* g1t = (_Float16*)carve((size_t)192 * 1024 * 2);  // Npad 160->192
    _Float16* g2t = (_Float16*)carve((size_t)1024 * 192 * 2);  // Kpad 160->192

    float* rbuf = (float*)carve(MC * 4);
    float* kbuf = (float*)carve(MC * 4);
    float* vbuf = (float*)carve(MC * 4);

    float* hw = (float*)carve((size_t)Mrows * 64 * 4);
    float* ha = (float*)carve((size_t)Mrows * 64 * 4);
    float* hv = (float*)carve((size_t)Mrows * 64 * 4);
    float* hg = (float*)carve((size_t)Mrows * 192 * 4);
    _Float16* hw16 = (_Float16*)carve((size_t)Mrows * 64 * 2);
    _Float16* ha16 = (_Float16*)carve((size_t)Mrows * 64 * 2);
    _Float16* hv16 = (_Float16*)carve((size_t)Mrows * 64 * 2);
    _Float16* hg16 = (_Float16*)carve((size_t)Mrows * 192 * 2);

    float* wpre = (float*)carve(MC * 4);
    float* apre = (float*)carve(MC * 4);
    float* vpre = (float*)carve(MC * 4);
    float* gbuf = (float*)carve(MC * 4);

    float* wdb = (float*)carve(MC * 4);
    float* nkk = (float*)carve(MC * 4);
    float* bkk = (float*)carve(MC * 4);
    float* ybuf = (float*)carve(MC * 4);
    _Float16* yg = (_Float16*)carve(MC * 2);

    // 1) token-shift mix -> f16
    mix_kernel<<<dim3((unsigned)(MC / 256)), dim3(256), 0, stream>>>(
        x, x_r, x_w, x_k, x_v, x_a, x_g, xr, xw, xk, xv, xa, xg);

    // 2) weight transpose/convert (Bt[n][k], zero-padded)
    auto wtrans = [&](const float* src, _Float16* dst, int K, int Ncols,
                      int Npad, int Kpad) {
        size_t tot = (size_t)Npad * Kpad;
        wtrans_kernel<<<dim3((unsigned)((tot + 255) / 256)), dim3(256), 0, stream>>>(
            src, dst, K, Ncols, Npad, Kpad);
    };
    wtrans(W_r, Wrt, 1024, 1024, 1024, 1024);
    wtrans(W_k, Wkt, 1024, 1024, 1024, 1024);
    wtrans(W_v, Wvt, 1024, 1024, 1024, 1024);
    wtrans(W_o, Wot, 1024, 1024, 1024, 1024);
    wtrans(w1, w1t, 1024,  64,  64, 1024);
    wtrans(w2, w2t,   64, 1024, 1024,  64);
    wtrans(a1, a1t, 1024,  64,  64, 1024);
    wtrans(a2, a2t,   64, 1024, 1024,  64);
    wtrans(v1, v1t, 1024,  32,  64, 1024);
    wtrans(v2, v2t,   32, 1024, 1024,  64);
    wtrans(g1, g1t, 1024, 160, 192, 1024);
    wtrans(g2, g2t,  160, 1024, 1024, 192);

    // 3) WMMA GEMMs (block tile 128x64)
    auto gemm = [&](const _Float16* A, const _Float16* Bt, float* C,
                    int Mm, int Nc, int K, int lda, int ldb, int ldc) {
        dim3 grid((unsigned)(Nc / 64), (unsigned)(Mm / 128));
        gemm_wmma_f16<<<grid, dim3(128), 0, stream>>>(A, Bt, C, K, lda, ldb, ldc);
    };

    // stage 1
    gemm(xr, Wrt, rbuf, Mrows, 1024, 1024, 1024, 1024, 1024);
    gemm(xk, Wkt, kbuf, Mrows, 1024, 1024, 1024, 1024, 1024);
    gemm(xv, Wvt, vbuf, Mrows, 1024, 1024, 1024, 1024, 1024);
    gemm(xw, w1t, hw, Mrows,  64, 1024, 1024, 1024,  64);
    gemm(xa, a1t, ha, Mrows,  64, 1024, 1024, 1024,  64);
    gemm(xv, v1t, hv, Mrows,  64, 1024, 1024, 1024,  64);
    gemm(xg, g1t, hg, Mrows, 192, 1024, 1024, 1024, 192);

    // 4) hidden activations -> f16
    act_kernel<<<dim3(Mrows * 64 / 256), dim3(256), 0, stream>>>(hw, hw16, Mrows * 64, 1);
    act_kernel<<<dim3(Mrows * 64 / 256), dim3(256), 0, stream>>>(ha, ha16, Mrows * 64, 0);
    act_kernel<<<dim3(Mrows * 64 / 256), dim3(256), 0, stream>>>(hv, hv16, Mrows * 64, 0);
    act_kernel<<<dim3(Mrows * 192 / 256), dim3(256), 0, stream>>>(hg, hg16, Mrows * 192, 2);

    // stage 2
    gemm(hw16, w2t, wpre, Mrows, 1024,  64,  64,  64, 1024);
    gemm(ha16, a2t, apre, Mrows, 1024,  64,  64,  64, 1024);
    gemm(hv16, v2t, vpre, Mrows, 1024,  64,  64,  64, 1024);
    gemm(hg16, g2t, gbuf, Mrows, 1024, 192, 192, 192, 1024);

    // 5) fused elementwise pre-scan
    prescan_kernel<<<dim3(Mrows), dim3(256), 0, stream>>>(
        wpre, apre, vpre, w0, a0, v0, k_k, k_a, v_first,
        kbuf, vbuf, wdb, nkk, bkk);

    // 6) sequential delta-rule scan (async LDS double-buffer)
    scan_kernel<<<dim3(Bb * Hh), dim3(Nn), 0, stream>>>(
        wdb, rbuf, kbuf, vbuf, nkk, bkk, ybuf);

    // 7) groupnorm + bonus + gate -> f16
    gn_post_kernel<<<dim3(Mrows * Hh), dim3(Nn), 0, stream>>>(
        ybuf, rbuf, kbuf, vbuf, gbuf, r_k, lnw, lnb, yg);

    // 8) output projection -> d_out[0 : M*C]
    gemm(yg, Wot, (float*)d_out, Mrows, 1024, 1024, 1024, 1024, 1024);

    // tuple output 2: v_first passthrough
    hipMemcpyAsync((float*)d_out + MC, v_first, MC * sizeof(float),
                   hipMemcpyDeviceToDevice, stream);
}
